// DeepseekV2MoE_72138270703855
// MI455X (gfx1250) — matrix operand order repaired and checked by reading, not compiled, not run on hardware
//
#include <hip/hip_runtime.h>
#include <hip/hip_bf16.h>
#include <math.h>

// Problem dims (fixed by reference)
#define E_  32
#define K_  6
#define T_  2048
#define H_  2048
#define I_  1408
#define SI_ 2816

#define IP_  (I_ + 8)    // padded LDS act row (bf16 elems), keeps 16B alignment
#define SIP_ (SI_ + 8)
#define PROW_ 17             // panel row stride in dwords (16 data + 1 TDM pad)
#define PSZ_  (32 * PROW_)   // dwords per 32x16 staged weight panel

typedef __attribute__((ext_vector_type(16))) __bf16 v16bf;
typedef __attribute__((ext_vector_type(8)))  __bf16 v8bf;
typedef __attribute__((ext_vector_type(8)))  float  v8f;
typedef __attribute__((ext_vector_type(4)))  unsigned int v4u;
typedef __attribute__((ext_vector_type(8)))  int v8i;
typedef __attribute__((ext_vector_type(4)))  int v4i;

__device__ __forceinline__ v8f wmma_bf16(v16bf a, v16bf b, v8f c) {
  // D = A(16x32 bf16) * B(32x16 bf16) + C(16x16 f32)
  return __builtin_amdgcn_wmma_f32_16x16x32_bf16(
      false, a, false, b, (short)0, c, false, false);
}

__device__ __forceinline__ float fast_silu_mul(float g, float u) {
  // silu(g)*u with v_rcp_f32 instead of IEEE divide
  return g * __builtin_amdgcn_rcpf(1.f + __expf(-g)) * u;
}

__device__ __forceinline__ void wait_dscnt0() {
  asm volatile("s_wait_dscnt 0x0" ::: "memory");
}

// ---- TDM: async-copy a 32x16 fp32 tile (global row stride = ldb elements)
// into LDS, padding 1 dword after every 16-dword row (pad_interval code 3,
// pad_amount code 0) so the staged panel has a 17-dword row stride.
// Descriptor per CDNA5 ISA ch.8; tracked on TENSORcnt.
__device__ __forceinline__ void tdm_load_panel(float* lds_dst,
                                               const float* gsrc, int ldb) {
  unsigned long long ga = (unsigned long long)(const void*)gsrc;
  unsigned int lo = (unsigned int)(unsigned long long)(void*)lds_dst;
  v4u g0;
  g0[0] = 1u;                                   // count=1 (valid user D#)
  g0[1] = lo;                                   // lds_addr
  g0[2] = (unsigned int)ga;                     // global_addr[31:0]
  g0[3] = ((unsigned int)(ga >> 32) & 0x1FFFFFFu) | 0x80000000u; // [56:32]|type=2
  v8i g1;
  g1[0] = (int)0x00D20000u;  // data_size=4B | pad_enable | interval=16dw | pad=1dw
  g1[1] = (int)(16u << 16);  // tensor_dim0 = 16  (bits 79:48, low half here)
  g1[2] = (int)(32u << 16);  // tensor_dim1 = 32  (bits 111:80, low half here)
  g1[3] = (int)(16u << 16);  // tile_dim0 = 16    (bits 127:112)
  g1[4] = 32;                // tile_dim1 = 32    (bits 143:128)
  g1[5] = ldb;               // tensor_dim0_stride[31:0]
  g1[6] = 0;
  g1[7] = 0;
  v4i z4 = {};
  v8i z8 = {};
  __builtin_amdgcn_tensor_load_to_lds(g0, g1, z4, z4, z8, 0);
}

// ---- A fragment: 16x32 bf16 from a global fp32 row (one M-row per lane) ----
__device__ __forceinline__ v16bf load_a_global(const float* __restrict__ row,
                                               int k0, int lane) {
  int kb = (lane >> 4) ? 8 : 0;
  const float* p = row + k0 + kb;
  float4 a0 = *(const float4*)(p + 0);
  float4 a1 = *(const float4*)(p + 4);
  float4 a2 = *(const float4*)(p + 16);
  float4 a3 = *(const float4*)(p + 20);
  v16bf A;
  A[0]  = (__bf16)a0.x; A[1]  = (__bf16)a0.y; A[2]  = (__bf16)a0.z; A[3]  = (__bf16)a0.w;
  A[4]  = (__bf16)a1.x; A[5]  = (__bf16)a1.y; A[6]  = (__bf16)a1.z; A[7]  = (__bf16)a1.w;
  A[8]  = (__bf16)a2.x; A[9]  = (__bf16)a2.y; A[10] = (__bf16)a2.z; A[11] = (__bf16)a2.w;
  A[12] = (__bf16)a3.x; A[13] = (__bf16)a3.y; A[14] = (__bf16)a3.z; A[15] = (__bf16)a3.w;
  return A;
}

// ---- A fragment from bf16 LDS act row (phase-2) ----
__device__ __forceinline__ v16bf load_a_lds(const __bf16* row, int k0, int lane) {
  int kb = (lane >> 4) ? 8 : 0;
  const v8bf* p = (const v8bf*)(row + k0 + kb);  // ds_load_b128
  v8bf lo = p[0];
  v8bf hi = p[2];
  v16bf A;
#pragma unroll
  for (int e = 0; e < 8; ++e) { A[e] = lo[e]; A[e + 8] = hi[e]; }
  return A;
}

// ---- B fragment: 32x16 bf16 from a TDM-staged fp32 LDS panel ----
__device__ __forceinline__ v16bf load_b_panel(const float* panel, int lane) {
  int n  = lane & 15;
  int kb = (lane >> 4) << 3;
  const float* p = panel + kb * PROW_ + n;
  v16bf B;
#pragma unroll
  for (int e = 0; e < 16; ++e) B[e] = (__bf16)p[((e & 7) + ((e >> 3) << 4)) * PROW_];
  return B;
}

// ============================== kernels ==================================

__global__ void zero_counts_kernel(int* __restrict__ cnt) {
  if (threadIdx.x < E_) cnt[threadIdx.x] = 0;
}

// One wave32 per token: 32 lanes = 32 experts. Softmax scoring, greedy top-6
// (tie-break lower index, matching lax.top_k), scatter to per-expert lists.
__global__ void moe_gate_kernel(const float* __restrict__ h,
                                const float* __restrict__ gate_w,
                                int* __restrict__ cnt,
                                int* __restrict__ tok,
                                float* __restrict__ twt) {
  int t = blockIdx.x;
  int lane = threadIdx.x;  // expert id
  const float* hp = h + (size_t)t * H_;
  const float* gp = gate_w + (size_t)lane * H_;
  float acc = 0.f;
  for (int i = 0; i < H_; i += 4) {
    float4 a = *(const float4*)(hp + i);
    float4 b = *(const float4*)(gp + i);
    acc += a.x * b.x + a.y * b.y + a.z * b.z + a.w * b.w;
  }
  float mx = acc;
  for (int off = 16; off; off >>= 1) mx = fmaxf(mx, __shfl_xor(mx, off, 32));
  float ex = __expf(acc - mx);
  float sm = ex;
  for (int off = 16; off; off >>= 1) sm += __shfl_xor(sm, off, 32);
  float prob = ex * __builtin_amdgcn_rcpf(sm);
  float val = prob;
  for (int k = 0; k < K_; ++k) {
    float bv = val; int bi = lane;
    for (int off = 16; off; off >>= 1) {
      float ov = __shfl_xor(bv, off, 32);
      int   oi = __shfl_xor(bi, off, 32);
      if (ov > bv || (ov == bv && oi < bi)) { bv = ov; bi = oi; }
    }
    if (lane == 0) {
      int slot = atomicAdd(&cnt[bi], 1);
      tok[bi * T_ + slot] = t;
      twt[bi * T_ + slot] = bv;
    }
    if (lane == bi) val = -INFINITY;
  }
}

// Shared expert: 16-token tile per block, 8 waves split N.
// Weight panels stream HBM->LDS via TDM, two slabs in flight, branch-free
// steady state (prologue / loop / peeled epilogue).
__global__ void __launch_bounds__(256)
shared_expert_kernel(const float* __restrict__ h,
                     const float* __restrict__ wg,
                     const float* __restrict__ wu,
                     const float* __restrict__ wd,
                     float* __restrict__ out) {
  __shared__ __align__(16) __bf16 act[16][SIP_];
  __shared__ __align__(16) float panels[8][2][2][PSZ_];  // [wave][buf][g|u]
  int t0 = blockIdx.x * 16;
  int wave = threadIdx.x >> 5, lane = threadIdx.x & 31;
  int m = lane & 15;
  const float* arow = h + (size_t)(t0 + m) * H_;
  float* Pg[2] = {&panels[wave][0][0][0], &panels[wave][1][0][0]};
  float* Pu[2] = {&panels[wave][0][1][0], &panels[wave][1][1][0]};

  for (int nc = wave; nc < SI_ / 16; nc += 8) {
    int n0 = nc * 16;
    v8f accg = {}, accu = {};
    auto consume = [&](int buf, int k0) {
      v16bf A  = load_a_global(arow, k0, lane);
      v16bf Bg = load_b_panel(Pg[buf], lane);
      v16bf Bu = load_b_panel(Pu[buf], lane);
      accg = wmma_bf16(A, Bg, accg);
      accu = wmma_bf16(A, Bu, accu);
    };
    // prologue: slabs 0 and 1 in flight
    tdm_load_panel(Pg[0], wg + n0, SI_);
    tdm_load_panel(Pu[0], wu + n0, SI_);
    tdm_load_panel(Pg[1], wg + (size_t)32 * SI_ + n0, SI_);
    tdm_load_panel(Pu[1], wu + (size_t)32 * SI_ + n0, SI_);
    int buf = 0;
    for (int k0 = 0; k0 < H_ - 64; k0 += 32) {
      __builtin_amdgcn_s_wait_tensorcnt(2);  // slab k0 resident
      consume(buf, k0);
      wait_dscnt0();                         // panel reads retired before reuse
      tdm_load_panel(Pg[buf], wg + (size_t)(k0 + 64) * SI_ + n0, SI_);
      tdm_load_panel(Pu[buf], wu + (size_t)(k0 + 64) * SI_ + n0, SI_);
      buf ^= 1;
    }
    __builtin_amdgcn_s_wait_tensorcnt(2);
    consume(buf, H_ - 64);
    buf ^= 1;
    __builtin_amdgcn_s_wait_tensorcnt(0);
    consume(buf, H_ - 32);

    int col = lane & 15;
    int rb  = (lane >> 4) ? 8 : 0;
#pragma unroll
    for (int r = 0; r < 8; ++r)
      act[rb + r][n0 + col] = (__bf16)fast_silu_mul(accg[r], accu[r]);
  }
  __syncthreads();
  for (int nc = wave; nc < H_ / 16; nc += 8) {
    int n0 = nc * 16;
    v8f acc = {};
    const __bf16* arow2 = &act[m][0];
    auto consume = [&](int buf, int k0) {
      v16bf A = load_a_lds(arow2, k0, lane);
      v16bf B = load_b_panel(Pg[buf], lane);
      acc = wmma_bf16(A, B, acc);
    };
    tdm_load_panel(Pg[0], wd + n0, H_);
    tdm_load_panel(Pg[1], wd + (size_t)32 * H_ + n0, H_);
    int buf = 0;
    for (int k0 = 0; k0 < SI_ - 64; k0 += 32) {
      __builtin_amdgcn_s_wait_tensorcnt(1);
      consume(buf, k0);
      wait_dscnt0();
      tdm_load_panel(Pg[buf], wd + (size_t)(k0 + 64) * H_ + n0, H_);
      buf ^= 1;
    }
    __builtin_amdgcn_s_wait_tensorcnt(1);
    consume(buf, SI_ - 64);
    buf ^= 1;
    __builtin_amdgcn_s_wait_tensorcnt(0);
    consume(buf, SI_ - 32);

    int col = lane & 15;
    int rb  = (lane >> 4) ? 8 : 0;
#pragma unroll
    for (int r = 0; r < 8; ++r)
      out[(size_t)(t0 + rb + r) * H_ + n0 + col] = acc[r];
  }
}

// Routed experts: block = (expert, 32-token tile of its gathered list).
__global__ void __launch_bounds__(256)
routed_expert_kernel(const float* __restrict__ h,
                     const float* __restrict__ w_gate,
                     const float* __restrict__ w_up,
                     const float* __restrict__ w_down,
                     const int* __restrict__ cnt,
                     const int* __restrict__ tok,
                     const float* __restrict__ twt,
                     float* __restrict__ out) {
  const int TILES = T_ / 32;  // 64
  int e    = blockIdx.x / TILES;
  int tile = blockIdx.x % TILES;
  int n = cnt[e];
  if (tile * 32 >= n) return;

  __shared__ __align__(16) __bf16 act[32][IP_];
  __shared__ __align__(16) float panels[8][2][2][PSZ_];
  __shared__ int   stok[32];
  __shared__ float swt[32];
  if (threadIdx.x < 32) {
    int idx  = tile * 32 + (int)threadIdx.x;
    int cidx = idx < n ? idx : (n - 1);
    stok[threadIdx.x] = tok[e * T_ + cidx];
    swt[threadIdx.x]  = idx < n ? twt[e * T_ + cidx] : 0.f;  // padded rows add 0
  }
  __syncthreads();

  int wave = threadIdx.x >> 5, lane = threadIdx.x & 31;
  int sub = wave & 1;   // 16-row subtile
  int nw  = wave >> 1;  // 4-way N split
  int m = lane & 15;
  const float* arow = h + (size_t)stok[sub * 16 + m] * H_;
  const float* wg = w_gate + (size_t)e * H_ * I_;
  const float* wu = w_up   + (size_t)e * H_ * I_;
  const float* wd = w_down + (size_t)e * I_ * H_;
  float* Pg[2] = {&panels[wave][0][0][0], &panels[wave][1][0][0]};
  float* Pu[2] = {&panels[wave][0][1][0], &panels[wave][1][1][0]};

  for (int nc = nw; nc < I_ / 16; nc += 4) {
    int n0 = nc * 16;
    v8f accg = {}, accu = {};
    auto consume = [&](int buf, int k0) {
      v16bf A  = load_a_global(arow, k0, lane);
      v16bf Bg = load_b_panel(Pg[buf], lane);
      v16bf Bu = load_b_panel(Pu[buf], lane);
      accg = wmma_bf16(A, Bg, accg);
      accu = wmma_bf16(A, Bu, accu);
    };
    tdm_load_panel(Pg[0], wg + n0, I_);
    tdm_load_panel(Pu[0], wu + n0, I_);
    tdm_load_panel(Pg[1], wg + (size_t)32 * I_ + n0, I_);
    tdm_load_panel(Pu[1], wu + (size_t)32 * I_ + n0, I_);
    int buf = 0;
    for (int k0 = 0; k0 < H_ - 64; k0 += 32) {
      __builtin_amdgcn_s_wait_tensorcnt(2);
      consume(buf, k0);
      wait_dscnt0();
      tdm_load_panel(Pg[buf], wg + (size_t)(k0 + 64) * I_ + n0, I_);
      tdm_load_panel(Pu[buf], wu + (size_t)(k0 + 64) * I_ + n0, I_);
      buf ^= 1;
    }
    __builtin_amdgcn_s_wait_tensorcnt(2);
    consume(buf, H_ - 64);
    buf ^= 1;
    __builtin_amdgcn_s_wait_tensorcnt(0);
    consume(buf, H_ - 32);

    int col = lane & 15;
    int rb  = sub * 16 + ((lane >> 4) ? 8 : 0);
#pragma unroll
    for (int r = 0; r < 8; ++r)
      act[rb + r][n0 + col] = (__bf16)fast_silu_mul(accg[r], accu[r]);
  }
  __syncthreads();
  for (int nc = nw; nc < H_ / 16; nc += 4) {
    int n0 = nc * 16;
    v8f acc = {};
    const __bf16* arow2 = &act[sub * 16 + m][0];
    auto consume = [&](int buf, int k0) {
      v16bf A = load_a_lds(arow2, k0, lane);
      v16bf B = load_b_panel(Pg[buf], lane);
      acc = wmma_bf16(A, B, acc);
    };
    tdm_load_panel(Pg[0], wd + n0, H_);
    tdm_load_panel(Pg[1], wd + (size_t)32 * H_ + n0, H_);
    int buf = 0;
    for (int k0 = 0; k0 < I_ - 64; k0 += 32) {
      __builtin_amdgcn_s_wait_tensorcnt(1);
      consume(buf, k0);
      wait_dscnt0();
      tdm_load_panel(Pg[buf], wd + (size_t)(k0 + 64) * H_ + n0, H_);
      buf ^= 1;
    }
    __builtin_amdgcn_s_wait_tensorcnt(1);
    consume(buf, I_ - 64);
    buf ^= 1;
    __builtin_amdgcn_s_wait_tensorcnt(0);
    consume(buf, I_ - 32);

    int col = lane & 15;
    int rb  = sub * 16 + ((lane >> 4) ? 8 : 0);
#pragma unroll
    for (int r = 0; r < 8; ++r) {
      int row = rb + r;
      atomicAdd(&out[(size_t)stok[row] * H_ + n0 + col], acc[r] * swt[row]);
    }
  }
}

// ============================== launcher =================================

extern "C" void kernel_launch(void* const* d_in, const int* in_sizes, int n_in,
                              void* d_out, int out_size, void* d_ws, size_t ws_size,
                              hipStream_t stream) {
  const float* h      = (const float*)d_in[0];
  const float* gate_w = (const float*)d_in[1];
  const float* w_gate = (const float*)d_in[2];
  const float* w_up   = (const float*)d_in[3];
  const float* w_down = (const float*)d_in[4];
  const float* ws_g   = (const float*)d_in[5];
  const float* ws_u   = (const float*)d_in[6];
  const float* ws_d   = (const float*)d_in[7];
  float* out = (float*)d_out;

  // workspace: [cnt: E ints][tok: E*T ints][twt: E*T floats] (~512KB)
  char*  ws  = (char*)d_ws;
  int*   cnt = (int*)ws;
  int*   tok = (int*)(ws + 256);
  float* twt = (float*)(ws + 256 + (size_t)E_ * T_ * sizeof(int));

  zero_counts_kernel<<<1, 32, 0, stream>>>(cnt);
  moe_gate_kernel<<<T_, 32, 0, stream>>>(h, gate_w, cnt, tok, twt);
  shared_expert_kernel<<<T_ / 16, 256, 0, stream>>>(h, ws_g, ws_u, ws_d, out);
  routed_expert_kernel<<<E_ * (T_ / 32), 256, 0, stream>>>(h, w_gate, w_up, w_down,
                                                           cnt, tok, twt, out);
}
